// QuBA_8950711844997
// MI455X (gfx1250) — matrix operand order repaired and checked by reading, not compiled, not run on hardware
//
#include <hip/hip_runtime.h>
#include <hip/hip_bf16.h>
#include <stdint.h>

#if __has_include(<hip/amd_detail/amd_gfx1250_TDM.h>)
#define TDM_6ARG 1
#else
#define TDM_6ARG 0
#endif

// Problem constants (match reference)
#define Nn      20000
#define Ee      320000
#define Ff      128
#define NIn     64
#define TEn     128
#define HEADSn  4
#define HDn     32
#define MSn     256
#define NOn     64
#define ITERSn  4
#define EPSf    1e-5f

typedef __attribute__((ext_vector_type(16))) __bf16        v16bf;
typedef __attribute__((ext_vector_type(8)))  float         v8f;
typedef __attribute__((ext_vector_type(4)))  unsigned int  u32x4;
typedef __attribute__((ext_vector_type(8)))  int           i32x8;
typedef __attribute__((ext_vector_type(4)))  int           i32x4;

union Frag16 { v16bf v; u32x4 q[2]; };

__device__ __forceinline__ float sigmf_(float x) { return 1.f / (1.f + __expf(-x)); }

// ---------------------------------------------------------------------------
// TDM: DMA a [64 x Kdim] bf16 strip of W (row stride = ldw elements) into LDS.
// D# packing per CDNA5 ISA:
//   group0: count=1 | lds_addr | global_addr[56:0] | type=2
//   group1: data_size=1 (2B), tensor_dim0=ldw, tensor_dim1=64,
//           tile_dim0=Kdim, tile_dim1=64, tensor_dim0_stride=ldw
// ---------------------------------------------------------------------------
__device__ __forceinline__ void tdm_load_W_strip(const __hip_bfloat16* gW,
                                                 unsigned lds_addr,
                                                 int Kdim, int ldw)
{
    unsigned long long ga = (unsigned long long)(uintptr_t)gW;
    unsigned dim0 = (unsigned)ldw;
    unsigned dim1 = 64u;
    unsigned long long s0 = (unsigned long long)ldw;   // dim0 stride (elements)

    u32x4 g0;
    g0[0] = 1u;                                        // count=1 (valid), user mode
    g0[1] = lds_addr;                                  // LDS byte address
    g0[2] = (unsigned)(ga & 0xFFFFFFFFu);              // global_addr[31:0]
    g0[3] = (unsigned)((ga >> 32) & 0x01FFFFFFu)       // global_addr[56:32]
          | (2u << 30);                                // type=2 ("image")

    i32x8 g1;
    g1[0] = (int)(1u << 16);                           // data_size=1 (2 bytes)
    g1[1] = (int)((dim0 & 0xFFFFu) << 16);             // tensor_dim0[15:0]
    g1[2] = (int)(((dim0 >> 16) & 0xFFFFu) | ((dim1 & 0xFFFFu) << 16));
    g1[3] = (int)(((dim1 >> 16) & 0xFFFFu) | (((unsigned)Kdim & 0xFFFFu) << 16)); // tile_dim0
    g1[4] = (int)(64u);                                // tile_dim1=64, tile_dim2=0
    g1[5] = (int)(s0 & 0xFFFFFFFFu);                   // tensor_dim0_stride[31:0]
    g1[6] = (int)((s0 >> 32) & 0xFFFFu);               // stride[47:32], dim1_stride=0
    g1[7] = 0;

    i32x4 z4 = {0, 0, 0, 0};
#if TDM_6ARG
    i32x8 z8 = {0, 0, 0, 0, 0, 0, 0, 0};
    __builtin_amdgcn_tensor_load_to_lds(g0, g1, z4, z4, z8, 0);
#else
    __builtin_amdgcn_tensor_load_to_lds(g0, g1, z4, z4, 0);
#endif
}

// ---------------------------------------------------------------------------
// WMMA GEMM:  C[M,Nout] = A[M,K](bf16) * W[Nout,K](bf16)^T  (+ bias) (+= if acc)
// Block: 256 thr = 8 waves; block tile = 128 rows x 64 cols.
// W strip staged in LDS once per block via TDM; all 8 waves read B fragments
// with ds_load_b128 (8x global-traffic reduction on W).
// A fragments double-buffered across 64-wide K steps (Kdim % 64 == 0 for all
// call sites) so A loads write the idle register pair -- no VALU copies, no
// WMMA->VALU WAR hazard NOPs.
// ---------------------------------------------------------------------------
__global__ __launch_bounds__(256)
void k_gemm_bf16(const __hip_bfloat16* __restrict__ A, int lda,
                 const __hip_bfloat16* __restrict__ W, int ldw,
                 const float* __restrict__ bias,
                 float* __restrict__ C, int ldc,
                 int M, int Nout, int Kdim, int accumulate)
{
    __shared__ alignas(16) unsigned short sW[64 * 256];   // up to 32 KB

    const int wave = threadIdx.x >> 5;
    const int lane = threadIdx.x & 31;
    const int half = lane >> 4;
    const int l16  = lane & 15;
    const int rowBase = (blockIdx.x * 8 + wave) * 16;
    const int colBase = blockIdx.y * 64;
    const bool active = rowBase < M;                  // wave-uniform

    // stage W[colBase .. colBase+63][0 .. Kdim) into LDS via Tensor Data Mover
    if (wave == 0) {
        tdm_load_W_strip(W + (size_t)colBase * ldw,
                         (unsigned)(uintptr_t)&sW[0], Kdim, ldw);
        __builtin_amdgcn_s_wait_tensorcnt(0);
    }
    __syncthreads();

    if (active) {
        v8f acc[4];
#pragma unroll
        for (int j = 0; j < 4; ++j) {
#pragma unroll
            for (int v = 0; v < 8; ++v) {
                int row = rowBase + half * 8 + v;
                int col = colBase + j * 16 + l16;
                acc[j][v] = accumulate ? C[(size_t)row * ldc + col] : 0.f;
            }
        }

        const __hip_bfloat16* arow = A + (size_t)(rowBase + l16) * lda + half * 8;
        Frag16 a0, a1;
        a0.q[0] = *reinterpret_cast<const u32x4*>(arow);
        a0.q[1] = *reinterpret_cast<const u32x4*>(arow + 16);

        for (int k = 0; k < Kdim; k += 64) {
            // prefetch A fragment for the second half-step (k+32)
            {
                const __hip_bfloat16* an = arow + (k + 32);
                a1.q[0] = *reinterpret_cast<const u32x4*>(an);
                a1.q[1] = *reinterpret_cast<const u32x4*>(an + 16);
            }
#pragma unroll
            for (int j = 0; j < 4; ++j) {
                Frag16 b;
                const unsigned short* bp =
                    &sW[(size_t)(j * 16 + l16) * Kdim + k + half * 8];
                b.q[0] = *reinterpret_cast<const u32x4*>(bp);
                b.q[1] = *reinterpret_cast<const u32x4*>(bp + 16);
                acc[j] = __builtin_amdgcn_wmma_f32_16x16x32_bf16(
                    false, a0.v, false, b.v, (short)0, acc[j], false, false);
            }
            // prefetch A fragment for the next 64-step (k+64)
            if (k + 64 < Kdim) {
                const __hip_bfloat16* an = arow + (k + 64);
                a0.q[0] = *reinterpret_cast<const u32x4*>(an);
                a0.q[1] = *reinterpret_cast<const u32x4*>(an + 16);
            }
#pragma unroll
            for (int j = 0; j < 4; ++j) {
                Frag16 b;
                const unsigned short* bp =
                    &sW[(size_t)(j * 16 + l16) * Kdim + (k + 32) + half * 8];
                b.q[0] = *reinterpret_cast<const u32x4*>(bp);
                b.q[1] = *reinterpret_cast<const u32x4*>(bp + 16);
                acc[j] = __builtin_amdgcn_wmma_f32_16x16x32_bf16(
                    false, a1.v, false, b.v, (short)0, acc[j], false, false);
            }
        }

#pragma unroll
        for (int j = 0; j < 4; ++j) {
            int col = colBase + j * 16 + l16;
            float bb = bias ? bias[col] : 0.f;
#pragma unroll
            for (int v = 0; v < 8; ++v) {
                int row = rowBase + half * 8 + v;
                C[(size_t)row * ldc + col] = acc[j][v] + bb;
            }
        }
    }
}

// ---------------------------------------------------------------------------
// Elementwise / helper kernels
// ---------------------------------------------------------------------------
__global__ void k_fill_f32(float* p, float v, int n) {
    int i = blockIdx.x * blockDim.x + threadIdx.x;
    if (i < n) p[i] = v;
}
__global__ void k_fill_i32(int* p, int v, int n) {
    int i = blockIdx.x * blockDim.x + threadIdx.x;
    if (i < n) p[i] = v;
}
__global__ void k_convert_bf16(const float* __restrict__ in,
                               __hip_bfloat16* __restrict__ out, int n) {
    int i = blockIdx.x * blockDim.x + threadIdx.x;
    if (i < n) out[i] = __float2bfloat16(in[i]);
}
__global__ void k_init_ns(const float* __restrict__ init_emb,
                          float* __restrict__ ns, __hip_bfloat16* __restrict__ nsb, int n) {
    int i = blockIdx.x * blockDim.x + threadIdx.x;
    if (i >= n) return;
    float v = init_emb[i & (Ff - 1)];
    ns[i] = v;
    nsb[i] = __float2bfloat16(v);
}

// Gather hs=ns[src], hd=ns[dst] -> msg [E,256] bf16 (cols 0..127 = hs, 128..255 = hd)
__global__ void k_gather_concat(const float* __restrict__ ns,
                                const int* __restrict__ src, const int* __restrict__ dst,
                                __hip_bfloat16* __restrict__ msg, int E)
{
    int t = blockIdx.x * blockDim.x + threadIdx.x;   // E * 64 quads
    if (t >= E * 64) return;
    int e = t >> 6;
    int c = (t & 63) * 4;                            // 0..252
    int node = (c < Ff) ? src[e] : dst[e];
    int cc = c & (Ff - 1);
    const float4 v = *reinterpret_cast<const float4*>(ns + (size_t)node * Ff + cc);
    __hip_bfloat16* o = msg + (size_t)e * (2 * Ff) + c;
    o[0] = __float2bfloat16(v.x);
    o[1] = __float2bfloat16(v.y);
    o[2] = __float2bfloat16(v.z);
    o[3] = __float2bfloat16(v.w);
}

// Column sums & sumsq for BatchNorm (batch dim = rows).  blockDim.x = 128.
#define STAT_ROWS 1024
__global__ void k_col_stats(const float* __restrict__ X, int M, int C,
                            float* __restrict__ stats)
{
    int col = blockIdx.x * blockDim.x + threadIdx.x;
    int r0 = blockIdx.y * STAT_ROWS;
    int r1 = r0 + STAT_ROWS; if (r1 > M) r1 = M;
    float s = 0.f, s2 = 0.f;
    for (int r = r0; r < r1; ++r) {
        float x = X[(size_t)r * C + col];
        s += x; s2 += x * x;
    }
    atomicAdd(&stats[col], s);
    atomicAdd(&stats[C + col], s2);
}

// y = (x-mean)*rsqrt(var+eps)*g + b ; optional relu ; optional f32/bf16 outputs
__global__ void k_bn_apply(const float* __restrict__ X, int M, int C,
                           const float* __restrict__ stats,
                           const float* __restrict__ g, const float* __restrict__ b,
                           int relu, float* __restrict__ outF,
                           __hip_bfloat16* __restrict__ outB)
{
    long long i = (long long)blockIdx.x * blockDim.x + threadIdx.x;
    long long total = (long long)M * C;
    if (i >= total) return;
    int c = (int)(i % C);
    float inv = 1.f / (float)M;
    float mean = stats[c] * inv;
    float var = stats[C + c] * inv - mean * mean;
    float y = (X[i] - mean) * rsqrtf(var + EPSf) * g[c] + b[c];
    if (relu) y = fmaxf(y, 0.f);
    if (outF) outF[i] = y;
    if (outB) outB[i] = __float2bfloat16(y);
}

__device__ __forceinline__ int enc_f(float f) {
    int i = __float_as_int(f);
    return i >= 0 ? i : (i ^ 0x7FFFFFFF);
}
__device__ __forceinline__ float dec_f(int e) {
    return __int_as_float(e >= 0 ? e : (e ^ 0x7FFFFFFF));
}

// per-(edge,head) attention logit + segment max (ordered-int atomicMax)
__global__ void k_score(const float* __restrict__ Qn, const float* __restrict__ Kn,
                        const float* __restrict__ tptr, const int* __restrict__ dst,
                        float* __restrict__ score, int* __restrict__ nmax, int E)
{
    int t = blockIdx.x * blockDim.x + threadIdx.x;
    if (t >= E * HEADSn) return;
    int e = t >> 2, h = t & 3;
    const float4* q = reinterpret_cast<const float4*>(Qn + (size_t)e * TEn + h * HDn);
    const float4* k = reinterpret_cast<const float4*>(Kn + (size_t)e * TEn + h * HDn);
    float dot = 0.f;
#pragma unroll
    for (int i = 0; i < 8; ++i) {
        float4 a = q[i], b = k[i];
        dot += a.x * b.x + a.y * b.y + a.z * b.z + a.w * b.w;
    }
    float temp = fminf(fmaxf(tptr[0], 0.5f), 5.0f);
    float s = dot / temp;
    s = (s > 0.f) ? s : 0.2f * s;                    // leaky_relu(0.2)
    score[t] = s;
    atomicMax(&nmax[dst[e] * HEADSn + h], enc_f(s));
}

__global__ void k_exp_sum(const float* __restrict__ score, const int* __restrict__ dst,
                          const int* __restrict__ nmax, float* __restrict__ w,
                          float* __restrict__ denom, int E)
{
    int t = blockIdx.x * blockDim.x + threadIdx.x;
    if (t >= E * HEADSn) return;
    int e = t >> 2, h = t & 3;
    float m = dec_f(nmax[dst[e] * HEADSn + h]);
    float ex = __expf(score[t] - m);
    w[t] = ex;
    atomicAdd(&denom[dst[e] * HEADSn + h], ex);
}

__global__ void k_wnorm(float* __restrict__ w, const float* __restrict__ denom,
                        const int* __restrict__ dst, int E)
{
    int t = blockIdx.x * blockDim.x + threadIdx.x;
    if (t >= E * HEADSn) return;
    int e = t >> 2, h = t & 3;
    w[t] = w[t] / denom[dst[e] * HEADSn + h];
}

// mraw[e][c] = w[e][c>>5] * raw[e][c]
__global__ void k_weight_raw(const float* __restrict__ raw, const float* __restrict__ w,
                             float* __restrict__ mraw, int E)
{
    int i = blockIdx.x * blockDim.x + threadIdx.x;
    if (i >= E * TEn) return;
    int e = i >> 7, c = i & (TEn - 1);
    mraw[i] = raw[i] * w[e * HEADSn + (c >> 5)];
}

// BN(messages) fused with scatter-add into agg[dst]
__global__ void k_bn_scatter(const float* __restrict__ mraw,
                             const float* __restrict__ stats,
                             const float* __restrict__ g, const float* __restrict__ b,
                             const int* __restrict__ dst, float* __restrict__ agg, int E)
{
    int i = blockIdx.x * blockDim.x + threadIdx.x;
    if (i >= E * TEn) return;
    int e = i >> 7, c = i & (TEn - 1);
    float inv = 1.f / (float)E;
    float mean = stats[c] * inv;
    float var = stats[TEn + c] * inv - mean * mean;
    float y = (mraw[i] - mean) * rsqrtf(var + EPSf) * g[c] + b[c];
    atomicAdd(&agg[(size_t)dst[e] * TEn + c], y);
}

// x = concat(agg[N,128], node_inputs[N,64]) -> bf16 [N,192]
__global__ void k_concat_x(const float* __restrict__ agg,
                           const float* __restrict__ node_inputs,
                           __hip_bfloat16* __restrict__ x, int n)
{
    int i = blockIdx.x * blockDim.x + threadIdx.x;
    if (i >= n * (TEn + NIn)) return;
    int r = i / (TEn + NIn), c = i % (TEn + NIn);
    float v = (c < TEn) ? agg[(size_t)r * TEn + c]
                        : node_inputs[(size_t)r * NIn + (c - TEn)];
    x[i] = __float2bfloat16(v);
}

// LSTM cell (c0=0, forget gate unused), residual:  ns += sig(o)*tanh(sig(i)*tanh(g))
__global__ void k_lstm(const float* __restrict__ gates, float* __restrict__ ns,
                       __hip_bfloat16* __restrict__ nsb, int n)
{
    int i = blockIdx.x * blockDim.x + threadIdx.x;
    if (i >= n * Ff) return;
    int r = i >> 7, c = i & (Ff - 1);
    const float* gr = gates + (size_t)r * (4 * Ff);
    float gi = gr[c];
    float gg = gr[2 * Ff + c];
    float go = gr[3 * Ff + c];
    float cv = sigmf_(gi) * tanhf(gg);
    float h = sigmf_(go) * tanhf(cv);
    float nv = ns[i] + h;
    ns[i] = nv;
    nsb[i] = __float2bfloat16(nv);
}

// ---------------------------------------------------------------------------
// Host-side orchestration
// ---------------------------------------------------------------------------
extern "C" void kernel_launch(void* const* d_in, const int* in_sizes, int n_in,
                              void* d_out, int out_size, void* d_ws, size_t ws_size,
                              hipStream_t stream)
{
    (void)in_sizes; (void)n_in; (void)out_size; (void)ws_size;

    const float* node_inputs = (const float*)d_in[0];
    const int*   src   = (const int*)d_in[1];
    const int*   dst   = (const int*)d_in[2];
    const float* init_emb = (const float*)d_in[3];
    const float* q_W = (const float*)d_in[4],  *q_b = (const float*)d_in[5];
    const float* q_g = (const float*)d_in[6],  *q_beta = (const float*)d_in[7];
    const float* k_W = (const float*)d_in[8],  *k_b = (const float*)d_in[9];
    const float* k_g = (const float*)d_in[10], *k_beta = (const float*)d_in[11];
    const float* temperature = (const float*)d_in[12];
    const float* m1_W = (const float*)d_in[13], *m1_b = (const float*)d_in[14];
    const float* m1_g = (const float*)d_in[15], *m1_bt = (const float*)d_in[16];
    const float* m2_W = (const float*)d_in[17], *m2_b = (const float*)d_in[18];
    const float* m2_g = (const float*)d_in[19], *m2_bt = (const float*)d_in[20];
    const float* m3_W = (const float*)d_in[21], *m3_b = (const float*)d_in[22];
    const float* m3_g = (const float*)d_in[23], *m3_bt = (const float*)d_in[24];
    const float* m4_W = (const float*)d_in[25], *m4_b = (const float*)d_in[26];
    const float* mn_g = (const float*)d_in[27], *mn_bt = (const float*)d_in[28];
    const float* W_ih = (const float*)d_in[29], *W_hh = (const float*)d_in[30];
    const float* b_ih = (const float*)d_in[31], *b_hh = (const float*)d_in[32];
    const float* fin_W = (const float*)d_in[33], *fin_b = (const float*)d_in[34];
    float* out = (float*)d_out;

    // --- carve workspace ---
    uint8_t* base = (uint8_t*)d_ws;
    size_t off = 0;
    auto carve = [&](size_t bytes) -> void* {
        void* p = base + off;
        off += (bytes + 255) & ~(size_t)255;
        return p;
    };
    __hip_bfloat16* msgB = (__hip_bfloat16*)carve((size_t)Ee * 2 * Ff * 2);
    float* Qraw  = (float*)carve((size_t)Ee * TEn * 4);
    float* Kraw  = (float*)carve((size_t)Ee * TEn * 4);
    float* hraw  = (float*)carve((size_t)Ee * MSn * 4);
    __hip_bfloat16* hB = (__hip_bfloat16*)carve((size_t)Ee * MSn * 2);
    float* scoreBuf = (float*)carve((size_t)Ee * HEADSn * 4);
    float* wBuf  = (float*)carve((size_t)Ee * HEADSn * 4);
    int*   nmax  = (int*)carve((size_t)Nn * HEADSn * 4);
    float* denom = (float*)carve((size_t)Nn * HEADSn * 4);
    float* statsBuf = (float*)carve(2 * 512 * 4);
    float* agg   = (float*)carve((size_t)Nn * TEn * 4);
    __hip_bfloat16* xB = (__hip_bfloat16*)carve((size_t)Nn * (TEn + NIn) * 2);
    float* gates = (float*)carve((size_t)Nn * 4 * Ff * 4);
    float* ns    = (float*)carve((size_t)Nn * Ff * 4);
    __hip_bfloat16* nsB = (__hip_bfloat16*)carve((size_t)Nn * Ff * 2);
    __hip_bfloat16* qWb  = (__hip_bfloat16*)carve((size_t)TEn * Ff * 2);
    __hip_bfloat16* kWb  = (__hip_bfloat16*)carve((size_t)TEn * Ff * 2);
    __hip_bfloat16* m1Wb = (__hip_bfloat16*)carve((size_t)MSn * 2 * Ff * 2);
    __hip_bfloat16* m2Wb = (__hip_bfloat16*)carve((size_t)MSn * MSn * 2);
    __hip_bfloat16* m3Wb = (__hip_bfloat16*)carve((size_t)MSn * MSn * 2);
    __hip_bfloat16* m4Wb = (__hip_bfloat16*)carve((size_t)TEn * MSn * 2);
    __hip_bfloat16* WihB = (__hip_bfloat16*)carve((size_t)4 * Ff * (TEn + NIn) * 2);
    __hip_bfloat16* WhhB = (__hip_bfloat16*)carve((size_t)4 * Ff * Ff * 2);
    __hip_bfloat16* finWb = (__hip_bfloat16*)carve((size_t)NOn * Ff * 2);
    float* raw128 = Qraw;   // reused after attention scores are done
    float* mraw   = Kraw;   // reused

    auto gblk = [](long long n) { return (unsigned)((n + 255) / 256); };
    auto cvt = [&](const float* s, __hip_bfloat16* d, int n) {
        k_convert_bf16<<<gblk(n), 256, 0, stream>>>(s, d, n);
    };
    auto gemm = [&](const __hip_bfloat16* A, int lda, const __hip_bfloat16* W, int ldw,
                    const float* bias, float* C, int ldc, int M, int Nout, int K, int acc) {
        dim3 g((unsigned)((M / 16 + 7) / 8), (unsigned)(Nout / 64));
        k_gemm_bf16<<<g, 256, 0, stream>>>(A, lda, W, ldw, bias, C, ldc, M, Nout, K, acc);
    };
    auto colstats = [&](const float* X, int M, int C) {
        k_fill_f32<<<gblk(2 * C), 256, 0, stream>>>(statsBuf, 0.f, 2 * C);
        dim3 g((unsigned)(C / 128), (unsigned)((M + STAT_ROWS - 1) / STAT_ROWS));
        k_col_stats<<<g, 128, 0, stream>>>(X, M, C, statsBuf);
    };
    auto bn = [&](const float* X, int M, int C, const float* g, const float* b,
                  int relu, float* oF, __hip_bfloat16* oB) {
        k_bn_apply<<<gblk((long long)M * C), 256, 0, stream>>>(X, M, C, statsBuf,
                                                               g, b, relu, oF, oB);
    };

    // One-time (per launch) weight conversion to bf16
    cvt(q_W, qWb, TEn * Ff);       cvt(k_W, kWb, TEn * Ff);
    cvt(m1_W, m1Wb, MSn * 2 * Ff); cvt(m2_W, m2Wb, MSn * MSn);
    cvt(m3_W, m3Wb, MSn * MSn);    cvt(m4_W, m4Wb, TEn * MSn);
    cvt(W_ih, WihB, 4 * Ff * (TEn + NIn)); cvt(W_hh, WhhB, 4 * Ff * Ff);
    cvt(fin_W, finWb, NOn * Ff);

    // ns = broadcast(init_emb)
    k_init_ns<<<gblk((long long)Nn * Ff), 256, 0, stream>>>(init_emb, ns, nsB, Nn * Ff);

    const int NEG_INF_ENC = (int)0x807FFFFF;   // enc_f(-inf)

    for (int it = 0; it < ITERSn; ++it) {
        // gather hs||hd -> msgB [E,256] bf16
        k_gather_concat<<<gblk((long long)Ee * 64), 256, 0, stream>>>(ns, src, dst, msgB, Ee);

        // Q = bn(hs @ qW^T + qb) ; K = bn(hd @ kW^T + kb)
        gemm(msgB,       2 * Ff, qWb, Ff, q_b, Qraw, TEn, Ee, TEn, Ff, 0);
        gemm(msgB + Ff,  2 * Ff, kWb, Ff, k_b, Kraw, TEn, Ee, TEn, Ff, 0);
        colstats(Qraw, Ee, TEn); bn(Qraw, Ee, TEn, q_g, q_beta, 0, Qraw, nullptr);
        colstats(Kraw, Ee, TEn); bn(Kraw, Ee, TEn, k_g, k_beta, 0, Kraw, nullptr);

        // attention: score -> segment max -> exp -> segment sum -> normalize
        k_fill_i32<<<gblk(Nn * HEADSn), 256, 0, stream>>>(nmax, NEG_INF_ENC, Nn * HEADSn);
        k_score<<<gblk((long long)Ee * HEADSn), 256, 0, stream>>>(Qraw, Kraw, temperature,
                                                                  dst, scoreBuf, nmax, Ee);
        k_fill_f32<<<gblk(Nn * HEADSn), 256, 0, stream>>>(denom, 0.f, Nn * HEADSn);
        k_exp_sum<<<gblk((long long)Ee * HEADSn), 256, 0, stream>>>(scoreBuf, dst, nmax,
                                                                    wBuf, denom, Ee);
        k_wnorm<<<gblk((long long)Ee * HEADSn), 256, 0, stream>>>(wBuf, denom, dst, Ee);

        // message MLP (256 -> 256 -> 256 -> 256 -> 128)
        gemm(msgB, 2 * Ff, m1Wb, 2 * Ff, m1_b, hraw, MSn, Ee, MSn, 2 * Ff, 0);
        colstats(hraw, Ee, MSn); bn(hraw, Ee, MSn, m1_g, m1_bt, 1, nullptr, hB);
        gemm(hB, MSn, m2Wb, MSn, m2_b, hraw, MSn, Ee, MSn, MSn, 0);
        colstats(hraw, Ee, MSn); bn(hraw, Ee, MSn, m2_g, m2_bt, 1, nullptr, hB);
        gemm(hB, MSn, m3Wb, MSn, m3_b, hraw, MSn, Ee, MSn, MSn, 0);
        colstats(hraw, Ee, MSn); bn(hraw, Ee, MSn, m3_g, m3_bt, 1, nullptr, hB);
        gemm(hB, MSn, m4Wb, MSn, m4_b, raw128, TEn, Ee, TEn, MSn, 0);

        // weight by attention, BN, scatter-add into agg
        k_weight_raw<<<gblk((long long)Ee * TEn), 256, 0, stream>>>(raw128, wBuf, mraw, Ee);
        colstats(mraw, Ee, TEn);
        k_fill_f32<<<gblk((long long)Nn * TEn), 256, 0, stream>>>(agg, 0.f, Nn * TEn);
        k_bn_scatter<<<gblk((long long)Ee * TEn), 256, 0, stream>>>(mraw, statsBuf,
                                                                    mn_g, mn_bt, dst, agg, Ee);

        // node LSTM update
        k_concat_x<<<gblk((long long)Nn * (TEn + NIn)), 256, 0, stream>>>(agg, node_inputs,
                                                                          xB, Nn);
        gemm(xB, TEn + NIn, WihB, TEn + NIn, b_ih, gates, 4 * Ff, Nn, 4 * Ff, TEn + NIn, 0);
        gemm(nsB, Ff, WhhB, Ff, b_hh, gates, 4 * Ff, Nn, 4 * Ff, Ff, 1);
        k_lstm<<<gblk((long long)Nn * Ff), 256, 0, stream>>>(gates, ns, nsB, Nn);

        // output head for this iteration
        gemm(nsB, Ff, finWb, Ff, fin_b, out + (size_t)it * Nn * NOn, NOn, Nn, NOn, Ff, 0);
    }
}
